// Network_7267084665614
// MI455X (gfx1250) — matrix operand order repaired and checked by reading, not compiled, not run on hardware
//
#include <hip/hip_runtime.h>

// ---------------------------------------------------------------------------
// 2-layer LSTM LM forward for MI455X (gfx1250, wave32, WMMA).
//
// Roofline: output projection writes 4096*50257*4B = 823MB -> ~35us at
// 23.3TB/s; it is store-bound, so f16 A/B + f32-accumulate WMMA is the right
// precision/path. The recurrence is latency-bound: 16 persistent WGs each own
// a 16-wide HID slice, recurrent weights live in LDS (96KB/WG, staged via
// async global->LDS DMA), h-state is exchanged through L2 with a device
// barrier (hw cluster barrier if the dispatch is a cluster, atomic-spin
// fallback otherwise).
// ---------------------------------------------------------------------------

typedef __attribute__((ext_vector_type(16))) _Float16 v16h;
typedef __attribute__((ext_vector_type(8)))  _Float16 v8h;
typedef __attribute__((ext_vector_type(8)))  float    v8f;
typedef int v4i_ __attribute__((vector_size(16)));   // b128 payload type

#define GLOBAL_AS __attribute__((address_space(1)))
#define LDS_AS    __attribute__((address_space(3)))

#define VOCAB     50257
#define VOCAB_PAD 50272     // 16*3142
#define NTILES    3142
#define EMB       256
#define HID       256
#define BATCH     16
#define TSEQ      256
#define ROWS      4096      // BATCH*TSEQ
#define NWG       16

#if defined(__HIP_DEVICE_COMPILE__) && \
    __has_builtin(__builtin_amdgcn_global_load_async_to_lds_b128)
#define HAVE_ASYNC_LDS 1
#endif

union AF { v16h v; v8h h[2]; };
union CF { v8f  v; float f[8]; };

__device__ __forceinline__ v16h ld_a(const _Float16* p) {
  // A 16x32 f16 tile: 16 contiguous halves per lane.
  AF a; a.h[0] = *(const v8h*)p; a.h[1] = *(const v8h*)(p + 8); return a.v;
}
__device__ __forceinline__ v16h ld_b(const _Float16* p) {
  // B 32x16 f16 tile (col-major source): 16 contiguous halves per lane.
  AF a; a.h[0] = *(const v8h*)p; a.h[1] = *(const v8h*)(p + 8); return a.v;
}
__device__ __forceinline__ v8f wmma_f16(v16h a, v16h b, v8f c) {
  return __builtin_amdgcn_wmma_f32_16x16x32_f16(false, a, false, b,
                                                (short)0, c, false, false);
}
__device__ __forceinline__ float sigf(float x) {
  return 1.0f / (1.0f + __expf(-x));
}

// Device-wide barrier: hardware cluster barrier when dispatched as a cluster
// (IB_STS2.CLUSTER_ID != 0), atomic-counter spin otherwise.
__device__ __forceinline__ void grid_barrier(unsigned* ctr, unsigned tgt,
                                             int incluster) {
  __threadfence();
  __syncthreads();
  if (incluster) {
    if (threadIdx.x == 0) __builtin_amdgcn_s_cluster_barrier();
  } else if (threadIdx.x == 0) {
    __hip_atomic_fetch_add(ctr, 1u, __ATOMIC_RELEASE, __HIP_MEMORY_SCOPE_AGENT);
    while (__hip_atomic_load(ctr, __ATOMIC_ACQUIRE, __HIP_MEMORY_SCOPE_AGENT) < tgt)
      __builtin_amdgcn_s_sleep(1);
  }
  __syncthreads();
}

// --------------------------- prep kernels ----------------------------------

// Zero sync counter + h double-buffers (must run every launch: graph replay).
__global__ void k_init(unsigned* ctr, _Float16* hbufs) {
  if (threadIdx.x == 0) *ctr = 0u;
  unsigned* p = (unsigned*)hbufs;          // 4 bufs * 4096 halves = 8192 u32
  for (int e = threadIdx.x; e < 8192; e += 256) p[e] = 0u;
}

// f32 -> f16 column-major transposes of W1x, W1h, W2.
__global__ void k_prep_w(const float* __restrict__ W1,
                         const float* __restrict__ W2,
                         _Float16* __restrict__ w1x_t,
                         _Float16* __restrict__ w1h_t,
                         _Float16* __restrict__ w2_t) {
  int idx = blockIdx.x * 256 + threadIdx.x;   // 1024 cols * 512 rows
  int col = idx & 1023, k = idx >> 10;
  w2_t[col * 512 + k] = (_Float16)W2[k * 1024 + col];
  if (k < 256) {
    w1x_t[col * 256 + k] = (_Float16)W1[k * 1024 + col];
    w1h_t[col * 256 + k] = (_Float16)W1[(k + 256) * 1024 + col];
  }
}

// Embedding gather -> f16 activations x[row = b*T+t][e].
__global__ void k_gather(const int* __restrict__ tokens,
                         const float* __restrict__ emb,
                         _Float16* __restrict__ x16) {
  int idx = blockIdx.x * 256 + threadIdx.x;   // ROWS*EMB
  int row = idx >> 8, e = idx & 255;
  x16[idx] = (_Float16)emb[(size_t)tokens[row] * EMB + e];
}

// Tiled transpose Wout [256][50257] f32 -> Wout_t [50272][256] f16 (padded 0).
__global__ void k_twout(const float* __restrict__ W, _Float16* __restrict__ Wt) {
  __shared__ float tile[16][17];
  int tx = threadIdx.x, ty = threadIdx.y;
  int colb = blockIdx.x * 16, kb = blockIdx.y * 16;
  int col = colb + tx;
  tile[ty][tx] = (col < VOCAB) ? W[(size_t)(kb + ty) * VOCAB + col] : 0.0f;
  __syncthreads();
  Wt[(size_t)(colb + ty) * 256 + kb + tx] = (_Float16)tile[tx][ty];
}

// ----------------- G1x = x @ W1x + b1  ([4096,256]@[256,1024]) -------------
__global__ void k_gemm_g1x(const _Float16* __restrict__ x16,
                           const _Float16* __restrict__ w1x_t,
                           const float* __restrict__ b1,
                           float* __restrict__ g1x) {
  const int tid = threadIdx.x, lane = tid & 31, wv = tid >> 5;
  const int ntile = blockIdx.x * 8 + wv;     // 0..63
  const int mtile = blockIdx.y;              // 0..255
  const int arow = lane & 15, hi = lane >> 4;
  const int ahalf = hi * 16, bko = hi * 16, dm = hi * 8, ncol = lane & 15;
  const int colb = ntile * 16, gcol = colb + ncol;

  CF acc;
  const float bv = b1[gcol];
#pragma unroll
  for (int r = 0; r < 8; ++r) acc.f[r] = bv;

  const _Float16* abase = x16 + (size_t)(mtile * 16 + arow) * 256;
  const _Float16* bbase = w1x_t + (size_t)(colb + ncol) * 256;
#pragma unroll
  for (int kk = 0; kk < 8; ++kk) {
    int K0 = kk * 32;
    acc.v = wmma_f16(ld_a(abase + K0 + ahalf), ld_b(bbase + K0 + bko), acc.v);
  }
#pragma unroll
  for (int r = 0; r < 8; ++r)
    g1x[(size_t)(mtile * 16 + dm + r) * 1024 + gcol] = acc.f[r];
}

// --------------------------- recurrent scan --------------------------------
// 16 persistent WGs x 128 threads (4 waves). WG `wg` owns HID slice
// [wg*16, wg*16+16); wave w computes the gate-block-w (i/j/f/o) WMMA N-tile.
__global__ void k_recur(const float* __restrict__ g1x,       // [4096][1024]
                        const _Float16* __restrict__ w1h_t,  // [1024][256] cm
                        const _Float16* __restrict__ w2_t,   // [1024][512] cm
                        const float* __restrict__ b2,
                        _Float16* __restrict__ hb,           // 4*[16][256] f16
                        _Float16* __restrict__ hs16,         // [4096][256]
                        unsigned* __restrict__ ctr) {
  __shared__ _Float16 s_w1[64 * 256];   // 32KB: this WG's W1h columns
  __shared__ _Float16 s_w2[64 * 512];   // 64KB: this WG's W2 columns
  __shared__ float    s_g[4 * 16 * 16]; //  4KB: i/j/f/o gate tiles

  const int tid = threadIdx.x;
  const int wg = blockIdx.x;
  const int lane = tid & 31, wv = tid >> 5;

  // Stage recurrent weight slices into LDS once. Prefer the CDNA5 async
  // global->LDS DMA path (ASYNCcnt) when the toolchain exposes it.
#ifdef HAVE_ASYNC_LDS
  for (int e = tid; e < 2048; e += 128) {            // W1h: 2048 x 16B
    int cl = e >> 5, k8 = e & 31;
    int gc = (cl >> 4) * 256 + wg * 16 + (cl & 15);
    __builtin_amdgcn_global_load_async_to_lds_b128(
        (GLOBAL_AS v4i_*)(w1h_t + (size_t)gc * 256 + k8 * 8),
        (LDS_AS v4i_*)(s_w1 + cl * 256 + k8 * 8),
        0, 0);
  }
  for (int e = tid; e < 4096; e += 128) {            // W2: 4096 x 16B
    int cl = e >> 6, k8 = e & 63;
    int gc = (cl >> 4) * 256 + wg * 16 + (cl & 15);
    __builtin_amdgcn_global_load_async_to_lds_b128(
        (GLOBAL_AS v4i_*)(w2_t + (size_t)gc * 512 + k8 * 8),
        (LDS_AS v4i_*)(s_w2 + cl * 512 + k8 * 8),
        0, 0);
  }
#if __has_builtin(__builtin_amdgcn_s_wait_asynccnt)
  __builtin_amdgcn_s_wait_asynccnt(0);
#else
  asm volatile("s_wait_asynccnt 0x0" ::: "memory");
#endif
#else
  for (int e = tid; e < 2048; e += 128) {
    int cl = e >> 5, k8 = e & 31;
    int gc = (cl >> 4) * 256 + wg * 16 + (cl & 15);
    ((uint4*)(s_w1 + cl * 256))[k8] = ((const uint4*)(w1h_t + (size_t)gc * 256))[k8];
  }
  for (int e = tid; e < 4096; e += 128) {
    int cl = e >> 6, k8 = e & 63;
    int gc = (cl >> 4) * 256 + wg * 16 + (cl & 15);
    ((uint4*)(s_w2 + cl * 512))[k8] = ((const uint4*)(w2_t + (size_t)gc * 512))[k8];
  }
#endif
  __syncthreads();

  const int arow = lane & 15, hi = lane >> 4;
  const int ahalf = hi * 16, bko = hi * 16, dm = hi * 8, ncol = lane & 15;
  const int gcol = wv * 256 + wg * 16 + ncol;

  const int eb = tid >> 4;          // elementwise batch 0..7 (and +8)
  const int eh = tid & 15;
  const int hslot = wg * 16 + eh;

  float c1a = 0.f, c1b = 0.f, c2a = 0.f, c2b = 0.f;
  const float bval2 = b2[gcol];
  unsigned tgt = 0;
  // IB_STS2[9:6] = CLUSTER_ID (0 => not dispatched as a cluster).
  const int incluster = (int)__builtin_amdgcn_s_getreg((3 << 11) | (6 << 6) | 28);

  for (int t = 0; t < TSEQ; ++t) {
    const int p = t & 1;
    _Float16*       h1c = hb + p * 4096;
    const _Float16* h1p = hb + (p ^ 1) * 4096;
    _Float16*       h2c = hb + 8192 + p * 4096;
    const _Float16* h2p = hb + 8192 + (p ^ 1) * 4096;

    // ---- layer 1: gates = G1x[t] + h1_prev @ W1h ----
    CF acc;
#pragma unroll
    for (int r = 0; r < 8; ++r)
      acc.f[r] = g1x[(size_t)((dm + r) * TSEQ + t) * 1024 + gcol];
#pragma unroll
    for (int kk = 0; kk < 8; ++kk) {
      int K0 = kk * 32;
      v16h a = ld_a(h1p + arow * 256 + K0 + ahalf);
      v16h b = ld_b(s_w1 + (wv * 16 + ncol) * 256 + K0 + bko);
      acc.v = wmma_f16(a, b, acc.v);
    }
#pragma unroll
    for (int r = 0; r < 8; ++r)
      s_g[wv * 256 + (dm + r) * 16 + ncol] = acc.f[r];

    if (t + 1 < TSEQ)   // pull next step's G1x slice toward the WGP
      __builtin_prefetch(g1x + (size_t)(arow * TSEQ + t + 1) * 1024 +
                         wv * 256 + wg * 16, 0, 1);
    __syncthreads();

    {
      float i0 = s_g[0 * 256 + eb * 16 + eh], j0 = s_g[1 * 256 + eb * 16 + eh];
      float f0 = s_g[2 * 256 + eb * 16 + eh], o0 = s_g[3 * 256 + eb * 16 + eh];
      c1a = c1a * sigf(f0 + 1.0f) + sigf(i0) * tanhf(j0);
      h1c[eb * 256 + hslot] = (_Float16)(tanhf(c1a) * sigf(o0));
      int b1i = eb + 8;
      float i1 = s_g[0 * 256 + b1i * 16 + eh], j1 = s_g[1 * 256 + b1i * 16 + eh];
      float f1 = s_g[2 * 256 + b1i * 16 + eh], o1 = s_g[3 * 256 + b1i * 16 + eh];
      c1b = c1b * sigf(f1 + 1.0f) + sigf(i1) * tanhf(j1);
      h1c[b1i * 256 + hslot] = (_Float16)(tanhf(c1b) * sigf(o1));
    }
    grid_barrier(ctr, tgt += NWG, incluster);   // publish h1(t)

    // ---- layer 2: gates = b2 + [h1(t), h2(t-1)] @ W2 ----
    CF acc2;
#pragma unroll
    for (int r = 0; r < 8; ++r) acc2.f[r] = bval2;
#pragma unroll
    for (int kk = 0; kk < 16; ++kk) {
      int K0 = kk * 32;
      v16h a = (K0 < 256) ? ld_a(h1c + arow * 256 + K0 + ahalf)
                          : ld_a(h2p + arow * 256 + (K0 - 256) + ahalf);
      v16h b = ld_b(s_w2 + (wv * 16 + ncol) * 512 + K0 + bko);
      acc2.v = wmma_f16(a, b, acc2.v);
    }
#pragma unroll
    for (int r = 0; r < 8; ++r)
      s_g[wv * 256 + (dm + r) * 16 + ncol] = acc2.f[r];
    __syncthreads();

    {
      float i0 = s_g[0 * 256 + eb * 16 + eh], j0 = s_g[1 * 256 + eb * 16 + eh];
      float f0 = s_g[2 * 256 + eb * 16 + eh], o0 = s_g[3 * 256 + eb * 16 + eh];
      c2a = c2a * sigf(f0 + 1.0f) + sigf(i0) * tanhf(j0);
      float h0 = tanhf(c2a) * sigf(o0);
      h2c[eb * 256 + hslot] = (_Float16)h0;
      hs16[(size_t)(eb * TSEQ + t) * 256 + hslot] = (_Float16)h0;
      int b1i = eb + 8;
      float i1 = s_g[0 * 256 + b1i * 16 + eh], j1 = s_g[1 * 256 + b1i * 16 + eh];
      float f1 = s_g[2 * 256 + b1i * 16 + eh], o1 = s_g[3 * 256 + b1i * 16 + eh];
      c2b = c2b * sigf(f1 + 1.0f) + sigf(i1) * tanhf(j1);
      float h1v = tanhf(c2b) * sigf(o1);
      h2c[b1i * 256 + hslot] = (_Float16)h1v;
      hs16[(size_t)(b1i * TSEQ + t) * 256 + hslot] = (_Float16)h1v;
    }
    grid_barrier(ctr, tgt += NWG, incluster);   // publish h2(t)
  }
}

// --------------- logits = hs @ Wout + bout  (store-bandwidth bound) --------
__global__ void k_proj(const _Float16* __restrict__ hs16,
                       const _Float16* __restrict__ wout_t,  // [50272][256] cm
                       const float* __restrict__ bout,
                       float* __restrict__ out) {
  const int tid = threadIdx.x, lane = tid & 31, wv = tid >> 5;
  const int ntile = blockIdx.x * 8 + wv;
  if (ntile >= NTILES) return;                 // wave-uniform
  const int arow = lane & 15, hi = lane >> 4;
  const int ahalf = hi * 16, bko = hi * 16, dm = hi * 8, ncol = lane & 15;
  const int colb = ntile * 16, lcol = colb + ncol;

  const float bv = (lcol < VOCAB) ? bout[lcol] : 0.0f;
  CF acc[4];
#pragma unroll
  for (int m = 0; m < 4; ++m)
#pragma unroll
    for (int r = 0; r < 8; ++r) acc[m].f[r] = bv;

  const int mb = blockIdx.y * 64;              // 4 M-tiles per wave: B reuse
  const _Float16* bbase = wout_t + (size_t)(colb + ncol) * 256;
#pragma unroll
  for (int kk = 0; kk < 8; ++kk) {
    int K0 = kk * 32;
    v16h b = ld_b(bbase + K0 + bko);
#pragma unroll
    for (int m = 0; m < 4; ++m) {
      v16h a = ld_a(hs16 + (size_t)(mb + m * 16 + arow) * 256 + K0 + ahalf);
      acc[m].v = wmma_f16(a, b, acc[m].v);
    }
  }
  if (lcol < VOCAB) {
#pragma unroll
    for (int m = 0; m < 4; ++m)
#pragma unroll
      for (int r = 0; r < 8; ++r)
        out[(size_t)(mb + m * 16 + dm + r) * VOCAB + lcol] = acc[m].f[r];
  }
}

// ------------------------------ launcher -----------------------------------
extern "C" void kernel_launch(void* const* d_in, const int* in_sizes, int n_in,
                              void* d_out, int out_size, void* d_ws, size_t ws_size,
                              hipStream_t stream) {
  const int*   tokens = (const int*)d_in[0];
  const float* emb    = (const float*)d_in[1];
  const float* W1     = (const float*)d_in[2];
  const float* b1     = (const float*)d_in[3];
  const float* W2     = (const float*)d_in[4];
  const float* b2     = (const float*)d_in[5];
  const float* Wout   = (const float*)d_in[6];
  const float* bout   = (const float*)d_in[7];
  float* out = (float*)d_out;

  auto al = [](size_t x) { return (x + 255) & ~(size_t)255; };
  char* w = (char*)d_ws;
  _Float16* x16    = (_Float16*)w; w += al((size_t)ROWS * EMB * 2);
  _Float16* w1x_t  = (_Float16*)w; w += al((size_t)1024 * 256 * 2);
  _Float16* w1h_t  = (_Float16*)w; w += al((size_t)1024 * 256 * 2);
  _Float16* w2_t   = (_Float16*)w; w += al((size_t)1024 * 512 * 2);
  _Float16* wout_t = (_Float16*)w; w += al((size_t)VOCAB_PAD * 256 * 2);
  float*    g1x    = (float*)w;    w += al((size_t)ROWS * 1024 * 4);
  _Float16* hs16   = (_Float16*)w; w += al((size_t)ROWS * 256 * 2);
  _Float16* hb     = (_Float16*)w; w += al((size_t)4 * 4096 * 2);
  unsigned* ctr    = (unsigned*)w; w += 256;

  k_init<<<1, 256, 0, stream>>>(ctr, hb);
  k_prep_w<<<2048, 256, 0, stream>>>(W1, W2, w1x_t, w1h_t, w2_t);
  k_gather<<<4096, 256, 0, stream>>>(tokens, emb, x16);
  k_twout<<<dim3(NTILES, 16), dim3(16, 16), 0, stream>>>(Wout, wout_t);
  k_gemm_g1x<<<dim3(8, 256), 256, 0, stream>>>(x16, w1x_t, b1, g1x);
  k_recur<<<NWG, 128, 0, stream>>>(g1x, w1h_t, w2_t, b2, hb, hs16, ctr);
  k_proj<<<dim3((NTILES + 7) / 8, 64), 256, 0, stream>>>(hs16, wout_t, bout, out);
}
